// NATSequenceEncoder_11922829213827
// MI455X (gfx1250) — matrix operand order repaired and checked by reading, not compiled, not run on hardware
//
#include <hip/hip_runtime.h>
#include <hip/hip_bf16.h>
#include <math.h>

// ---------------------------------------------------------------------------
// NATSequenceEncoder for MI455X (gfx1250, wave32).
//
// Key optimization: the reference output is out[:, :, -1] only, and the whole
// network is local (convs + sliding-window attention). The receptive field of
// the last position is ~40 trailing tokens at level 0, so we run the entire
// backbone on tail slabs: S = {48, 24, 16} tokens per level (halo margin
// absorbs left-edge clamping error). All token-wise GEMMs (qkv / proj / fc1 /
// fc2) run as f16 WMMA (v_wmma_f32_16x16x32_f16) with f32 accumulate and
// fused epilogues (bias, exact GELU, residual add).
// ---------------------------------------------------------------------------

typedef _Float16 v16h __attribute__((ext_vector_type(16)));
typedef _Float16 v8h  __attribute__((ext_vector_type(8)));
typedef float    v8f  __attribute__((ext_vector_type(8)));

// ---------------------------- f32 -> f16 conversion ------------------------
__global__ void k_f32_to_f16(const float* __restrict__ s, _Float16* __restrict__ d, int n) {
  int i = blockIdx.x * blockDim.x + threadIdx.x;
  if (i < n) d[i] = (_Float16)s[i];
}

// ---------------------------- WMMA GEMM ------------------------------------
// C[M,N] = A[M,K](f16, row-major) * W[N,K]^T(f16, row-major) + bias[N]
// MODE 0: store f16        MODE 1: store f16 after exact GELU
// MODE 2: store f32 = res[M,N] + C   (residual add; may alias res)
//
// Fragment layouts used here (contiguous k order, same for A and B):
//  A 16x32 f16: lanes 0-15 -> M=lane, K elems {kb+0..15};
//               lanes 16-31 -> M=lane-16, K elems {kb+16..31}.
//               => two contiguous b128 loads per lane from row-major A.
//  B 32x16 f16: lanes 0-15 -> N=lane, K=kb+0..15; lanes 16-31 -> K=kb+16..31.
//               => 16 contiguous K halves per lane == one 32B load of W row.
//  C/D 16x16 f32: vgpr r: lanes 0-15 -> (M=r, N=lane); lanes 16-31 -> (M=8+r).
template<int MODE>
__global__ __launch_bounds__(128) void k_gemm_wmma(
    const _Float16* __restrict__ A, const _Float16* __restrict__ W,
    const float* __restrict__ bias,
    _Float16* __restrict__ out16, float* __restrict__ out32,
    const float* __restrict__ res,
    int M, int N, int K)
{
  const int lane = threadIdx.x & 31;
  const int wv   = threadIdx.x >> 5;
  const int nT = N >> 4, mT = M >> 4;
  const int tile = blockIdx.x * 4 + wv;
  if (tile >= nT * mT) return;                 // wave-uniform exit: EXEC all-1 for WMMA
  const int mt = tile / nT, nt = tile % nT;
  const int hi = lane >> 4;                    // lane half
  const int l  = lane & 15;

  const _Float16* ar = A + (size_t)(mt * 16 + l) * K + hi * 16;
  const _Float16* wr = W + (size_t)(nt * 16 + l) * K + hi * 16;

  v8f c = {};
  for (int kb = 0; kb < K; kb += 32) {
    v8h a0 = *(const v8h*)(ar + kb);           // K halves {0..7} of this lane-half
    v8h a1 = *(const v8h*)(ar + kb + 8);       // K halves {8..15} of this lane-half
    v16h a = __builtin_shufflevector(a0, a1, 0,1,2,3,4,5,6,7,8,9,10,11,12,13,14,15);
    v16h b = *(const v16h*)(wr + kb);          // 16 contiguous K halves of W row n
    c = __builtin_amdgcn_wmma_f32_16x16x32_f16(
        /*neg_a=*/false, a, /*neg_b=*/false, b,
        /*c_mod=*/(short)0, c, /*reuse_a=*/false, /*reuse_b=*/false);
  }

  const int n = nt * 16 + l;
  const float bv = bias[n];
  #pragma unroll
  for (int r = 0; r < 8; ++r) {
    const int m = mt * 16 + hi * 8 + r;
    float v = c[r] + bv;
    const size_t off = (size_t)m * N + n;
    if (MODE == 0) {
      out16[off] = (_Float16)v;
    } else if (MODE == 1) {
      out16[off] = (_Float16)(0.5f * v * (1.0f + erff(v * 0.70710678118f)));
    } else {
      out32[off] = res[off] + v;
    }
  }
}

// ---------------------------- LayerNorm ------------------------------------
template<bool F16OUT>
__global__ void k_layernorm(const float* __restrict__ x, const float* __restrict__ g,
                            const float* __restrict__ b, _Float16* __restrict__ o16,
                            float* __restrict__ o32, int rows, int d)
{
  int r = blockIdx.x * blockDim.x + threadIdx.x;
  if (r >= rows) return;
  const float* xr = x + (size_t)r * d;
  float m = 0.f;
  for (int i = 0; i < d; ++i) m += xr[i];
  m /= (float)d;
  float v = 0.f;
  for (int i = 0; i < d; ++i) { float t = xr[i] - m; v += t * t; }
  v /= (float)d;
  const float inv = rsqrtf(v + 1e-5f);
  if (F16OUT) {
    _Float16* orow = o16 + (size_t)r * d;
    for (int i = 0; i < d; ++i) orow[i] = (_Float16)((xr[i] - m) * inv * g[i] + b[i]);
  } else {
    float* orow = o32 + (size_t)r * d;
    for (int i = 0; i < d; ++i) orow[i] = (xr[i] - m) * inv * g[i] + b[i];
  }
}

// ---------------------------- ConvTokenizer (tail slab) --------------------
// y[b, p, o] = bias[o] + sum_{c<8, dk<3} x[b, c, t-1+dk] * w[o, c, dk], t = T-S+p
__global__ void k_tokenizer(const float* __restrict__ x, const float* __restrict__ w,
                            const float* __restrict__ bias, float* __restrict__ y,
                            int B, int T, int S)
{
  int i = blockIdx.x * blockDim.x + threadIdx.x;
  int total = B * S * 32;
  if (i >= total) return;
  int o = i % 32, t2 = i / 32;
  int p = t2 % S, b = t2 / S;
  int t = T - S + p;
  float acc = bias[o];
  for (int c = 0; c < 8; ++c)
    for (int dk = 0; dk < 3; ++dk) {
      int tt = t - 1 + dk;
      if (tt < 0 || tt >= T) continue;               // zero pad
      acc += x[((size_t)b * 8 + c) * T + tt] * w[(o * 8 + c) * 3 + dk];
    }
  y[((size_t)(b * S + p)) * 32 + o] = acc;
}

// ---------------------------- NATTEN 1D attention (head_dim == 16) ---------
// qkv: [B*S, 3d] f16, o: [B*S, d] f16. Edge-clamped window, rpb[H, 2k-1].
__global__ void k_nat_attn(const _Float16* __restrict__ qkv, const float* __restrict__ rpb,
                           _Float16* __restrict__ o, int B, int S, int T, int d, int H, int ks)
{
  int i = blockIdx.x * blockDim.x + threadIdx.x;
  int total = B * S * H;
  if (i >= total) return;
  const int HD = 16;                          // d/H == 16 for all levels
  int h = i % H, t2 = i / H;
  int p = t2 % S, b = t2 / S;
  int t = T - S + p;
  int start = t - ks / 2;
  if (start < 0) start = 0;
  if (start > T - ks) start = T - ks;         // NATTEN clamp at right edge
  const float scale = 0.25f;                  // 16^-0.5
  float q[HD];
  const _Float16* qb = qkv + ((size_t)(b * S + p)) * 3 * d + h * HD;
  for (int e = 0; e < HD; ++e) q[e] = (float)qb[e] * scale;

  float lg[5];
  float mx = -1e30f;
  for (int wi = 0; wi < ks; ++wi) {
    int ja = start + wi;
    int j = ja - (T - S); if (j < 0) j = 0;   // slab-left clamp (halo margin)
    const _Float16* kb = qkv + ((size_t)(b * S + j)) * 3 * d + d + h * HD;
    float a = 0.f;
    for (int e = 0; e < HD; ++e) a += q[e] * (float)kb[e];
    int rel = ja - t + ks - 1;
    if (rel < 0) rel = 0;
    if (rel > 2 * ks - 2) rel = 2 * ks - 2;
    a += rpb[h * (2 * ks - 1) + rel];
    lg[wi] = a; mx = fmaxf(mx, a);
  }
  float den = 0.f;
  for (int wi = 0; wi < ks; ++wi) { lg[wi] = __expf(lg[wi] - mx); den += lg[wi]; }
  const float inv = 1.f / den;
  float acc[HD];
  for (int e = 0; e < HD; ++e) acc[e] = 0.f;
  for (int wi = 0; wi < ks; ++wi) {
    int ja = start + wi;
    int j = ja - (T - S); if (j < 0) j = 0;
    const _Float16* vb = qkv + ((size_t)(b * S + j)) * 3 * d + 2 * d + h * HD;
    float aw = lg[wi] * inv;
    for (int e = 0; e < HD; ++e) acc[e] += aw * (float)vb[e];
  }
  _Float16* ob = o + ((size_t)(b * S + p)) * d + h * HD;
  for (int e = 0; e < HD; ++e) ob[e] = (_Float16)acc[e];
}

// ---------------------------- stride-2 downsample conv ---------------------
// y[b, p', o] = sum_{dk,c} x[b, j, c] * w[o, c, dk],  j = 2p'+dk-1 + (S-2*Sn)
__global__ void k_ds_conv(const float* __restrict__ x, const float* __restrict__ w,
                          float* __restrict__ y, int B, int S, int Sn, int d)
{
  int i = blockIdx.x * blockDim.x + threadIdx.x;
  int d2 = 2 * d;
  int total = B * Sn * d2;
  if (i >= total) return;
  int o = i % d2, t2 = i / d2;
  int p = t2 % Sn, b = t2 / Sn;
  float acc = 0.f;
  for (int dk = 0; dk < 3; ++dk) {
    int j = 2 * p + dk - 1 + (S - 2 * Sn);
    if (j < 0) j = 0;                         // halo margin
    if (j > S - 1) j = S - 1;
    const float* xr = x + ((size_t)(b * S + j)) * d;
    const float* wr = w + ((size_t)o * d) * 3 + dk;
    for (int c = 0; c < d; ++c) acc += xr[c] * wr[c * 3];
  }
  y[((size_t)(b * Sn + p)) * d2 + o] = acc;
}

// ---------------------------- FPN lateral conv (k=3, pad=1) ----------------
__global__ void k_lat_conv(const _Float16* __restrict__ xin, const float* __restrict__ w,
                           const float* __restrict__ bias, float* __restrict__ y,
                           int B, int S, int Cin)
{
  int i = blockIdx.x * blockDim.x + threadIdx.x;
  int total = B * S * 128;
  if (i >= total) return;
  int o = i % 128, t2 = i / 128;
  int p = t2 % S, b = t2 / S;
  float acc = bias[o];
  for (int dk = 0; dk < 3; ++dk) {
    int j = p - 1 + dk;
    if (j >= S) continue;                     // true right-edge zero pad (slab end == T)
    if (j < 0) j = 0;                         // halo margin
    const _Float16* xr = xin + ((size_t)(b * S + j)) * Cin;
    const float* wr = w + (size_t)o * Cin * 3 + dk;
    for (int c = 0; c < Cin; ++c) acc += (float)xr[c] * wr[c * 3];
  }
  y[((size_t)(b * S + p)) * 128 + o] = acc;
}

// ---------------------------- FPN linear upsample + add --------------------
__global__ void k_fpn_upadd(float* __restrict__ lo, const float* __restrict__ hi,
                            int B, int Slo, int Shi, int Tlo, int Thi)
{
  int i = blockIdx.x * blockDim.x + threadIdx.x;
  int total = B * Slo * 128;
  if (i >= total) return;
  int c = i % 128, t2 = i / 128;
  int p = t2 % Slo, b = t2 / Slo;
  int t = Tlo - Slo + p;
  float pos = ((float)t + 0.5f) * ((float)Thi / (float)Tlo) - 0.5f;
  if (pos < 0.f) pos = 0.f;
  float mxp = (float)(Thi - 1);
  if (pos > mxp) pos = mxp;
  int i0 = (int)pos;
  int i1 = i0 + 1; if (i1 > Thi - 1) i1 = Thi - 1;
  float w = pos - (float)i0;
  int base = Thi - Shi;
  int j0 = i0 - base; if (j0 < 0) j0 = 0; if (j0 > Shi - 1) j0 = Shi - 1;
  int j1 = i1 - base; if (j1 < 0) j1 = 0; if (j1 > Shi - 1) j1 = Shi - 1;
  lo[((size_t)(b * Slo + p)) * 128 + c] +=
      hi[((size_t)(b * Shi + j0)) * 128 + c] * (1.f - w) +
      hi[((size_t)(b * Shi + j1)) * 128 + c] * w;
}

// ---------------------------- final FPN conv at t = T-1 --------------------
__global__ void k_fpn_final(const float* __restrict__ lat0, const float* __restrict__ w,
                            const float* __restrict__ bias, float* __restrict__ out,
                            int B, int S)
{
  int i = blockIdx.x * blockDim.x + threadIdx.x;
  int total = B * 128;
  if (i >= total) return;
  int o = i % 128, b = i / 128;
  float acc = bias[o];
  for (int kk = 0; kk < 2; ++kk) {            // kk==2 tap hits t==T: zero pad
    const float* xr = lat0 + ((size_t)(b * S + (S - 2 + kk))) * 128;
    const float* wr = w + (size_t)o * 128 * 3 + kk;
    for (int c = 0; c < 128; ++c) acc += xr[c] * wr[c * 3];
  }
  out[(size_t)b * 128 + o] = acc;
}

// ---------------------------------------------------------------------------
extern "C" void kernel_launch(void* const* d_in, const int* in_sizes, int n_in,
                              void* d_out, int out_size, void* d_ws, size_t ws_size,
                              hipStream_t stream)
{
  (void)in_sizes; (void)n_in; (void)out_size; (void)ws_size;
  const int B = 128;
  const int Dd[3] = {32, 64, 128};
  const int Hh[3] = {2, 4, 8};
  const int Ksz[3] = {3, 3, 5};
  const int Tt[3] = {4096, 2048, 1024};
  const int Ss[3] = {48, 24, 16};              // tail slabs (receptive field + halo)

  auto F = [&](int i) -> const float* { return (const float*)d_in[i]; };

  // deterministic workspace carve-out (~15 MB total)
  char* wp = (char*)d_ws;
  auto alloc = [&](size_t bytes) -> void* {
    void* p = (void*)wp;
    wp += (bytes + 255) & ~(size_t)255;
    return p;
  };
  _Float16* t16   = (_Float16*)alloc(262144 * 2);   // LN output, max B*S*d
  _Float16* qkv16 = (_Float16*)alloc(786432 * 2);   // qkv, max B*S*3d
  _Float16* o16   = (_Float16*)alloc(262144 * 2);   // attention out
  _Float16* mid16 = (_Float16*)alloc(786432 * 2);   // MLP hidden
  _Float16* wq16  = (_Float16*)alloc(49152 * 2);
  _Float16* wp16  = (_Float16*)alloc(16384 * 2);
  _Float16* w116  = (_Float16*)alloc(49152 * 2);
  _Float16* w216  = (_Float16*)alloc(49152 * 2);
  float* h[3]; _Float16* outs[3]; float* lat[3];
  for (int i = 0; i < 3; ++i) h[i]    = (float*)alloc((size_t)B * Ss[i] * Dd[i] * 4);
  for (int i = 0; i < 3; ++i) outs[i] = (_Float16*)alloc((size_t)B * Ss[i] * Dd[i] * 2);
  for (int i = 0; i < 3; ++i) lat[i]  = (float*)alloc((size_t)B * Ss[i] * 128 * 4);
  float* dsTmp = (float*)alloc(262144 * 4);

  auto cvt = [&](const float* s, _Float16* d, int n) {
    k_f32_to_f16<<<(n + 255) / 256, 256, 0, stream>>>(s, d, n);
  };
  auto gemm = [&](int mode, const _Float16* A, const _Float16* Wm, const float* bias,
                  _Float16* O16, float* O32, const float* R, int M, int N, int K) {
    int tiles = (M / 16) * (N / 16);
    dim3 g((tiles + 3) / 4), blk(128);
    if (mode == 0)      k_gemm_wmma<0><<<g, blk, 0, stream>>>(A, Wm, bias, O16, O32, R, M, N, K);
    else if (mode == 1) k_gemm_wmma<1><<<g, blk, 0, stream>>>(A, Wm, bias, O16, O32, R, M, N, K);
    else                k_gemm_wmma<2><<<g, blk, 0, stream>>>(A, Wm, bias, O16, O32, R, M, N, K);
  };

  // Tokenizer: x[B,8,T] -> h0[B,S0,32] (tail slab)
  {
    int total = B * Ss[0] * 32;
    k_tokenizer<<<(total + 255) / 256, 256, 0, stream>>>(F(0), F(1), F(2), h[0], B, Tt[0], Ss[0]);
  }

  int ix = 3;  // input index walker: levels after x, embed_w, embed_b
  for (int lv = 0; lv < 3; ++lv) {
    const int d = Dd[lv], S = Ss[lv], T = Tt[lv], H = Hh[lv], ks = Ksz[lv];
    const int M = B * S, hdn = 3 * d;
    for (int ly = 0; ly < 2; ++ly) {
      const float *n1g = F(ix), *n1b = F(ix + 1), *qkvw = F(ix + 2), *qkvb = F(ix + 3),
                  *rpb = F(ix + 4), *projw = F(ix + 5), *projb = F(ix + 6),
                  *n2g = F(ix + 7), *n2b = F(ix + 8), *fc1w = F(ix + 9), *fc1b = F(ix + 10),
                  *fc2w = F(ix + 11), *fc2b = F(ix + 12);
      ix += 13;
      cvt(qkvw, wq16, hdn * d);
      cvt(projw, wp16, d * d);
      cvt(fc1w, w116, hdn * d);
      cvt(fc2w, w216, d * hdn);
      // x + attn(LN1(x))
      k_layernorm<true><<<(M + 255) / 256, 256, 0, stream>>>(h[lv], n1g, n1b, t16, nullptr, M, d);
      gemm(0, t16, wq16, qkvb, qkv16, nullptr, nullptr, M, hdn, d);
      {
        int total = B * S * H;
        k_nat_attn<<<(total + 255) / 256, 256, 0, stream>>>(qkv16, rpb, o16, B, S, T, d, H, ks);
      }
      gemm(2, o16, wp16, projb, nullptr, h[lv], h[lv], M, d, d);   // in-place residual
      // x + fc2(gelu(fc1(LN2(x))))
      k_layernorm<true><<<(M + 255) / 256, 256, 0, stream>>>(h[lv], n2g, n2b, t16, nullptr, M, d);
      gemm(1, t16, w116, fc1b, mid16, nullptr, nullptr, M, hdn, d);
      gemm(2, mid16, w216, fc2b, nullptr, h[lv], h[lv], M, d, hdn);
    }
    const float *og = F(ix), *ob = F(ix + 1); ix += 2;
    k_layernorm<true><<<(M + 255) / 256, 256, 0, stream>>>(h[lv], og, ob, outs[lv], nullptr, M, d);
    if (lv < 2) {
      const float *dsw = F(ix), *dsg = F(ix + 1), *dsb = F(ix + 2); ix += 3;
      const int Sn = Ss[lv + 1], d2 = 2 * d, Mn = B * Sn;
      int total = Mn * d2;
      k_ds_conv<<<(total + 255) / 256, 256, 0, stream>>>(h[lv], dsw, dsTmp, B, S, Sn, d);
      k_layernorm<false><<<(Mn + 255) / 256, 256, 0, stream>>>(dsTmp, dsg, dsb, nullptr, h[lv + 1], Mn, d2);
    }
  }

  // FPN: lateral convs -> top-down upsample+add -> final conv at t = T-1
  for (int lv = 0; lv < 3; ++lv) {
    int total = B * Ss[lv] * 128;
    k_lat_conv<<<(total + 255) / 256, 256, 0, stream>>>(outs[lv], F(93 + lv), F(96 + lv),
                                                        lat[lv], B, Ss[lv], Dd[lv]);
  }
  {
    int total = B * Ss[1] * 128;
    k_fpn_upadd<<<(total + 255) / 256, 256, 0, stream>>>(lat[1], lat[2], B, Ss[1], Ss[2], Tt[1], Tt[2]);
    total = B * Ss[0] * 128;
    k_fpn_upadd<<<(total + 255) / 256, 256, 0, stream>>>(lat[0], lat[1], B, Ss[0], Ss[1], Tt[0], Tt[1]);
    total = B * 128;
    k_fpn_final<<<(total + 255) / 256, 256, 0, stream>>>(lat[0], F(99), F(100), (float*)d_out, B, Ss[0]);
  }
}
